// SequenceEncoder_50792283242658
// MI455X (gfx1250) — matrix operand order repaired and checked by reading, not compiled, not run on hardware
//
#include <hip/hip_runtime.h>
#include <hip/hip_bf16.h>

// ---------------------------------------------------------------------------
// SequenceEncoder (attention + pair-bias + FFN block) for gfx1250 / MI455X.
// All GEMMs via v_wmma_f32_16x16x32_f16 (f16 in, f32 accumulate).
// B=8 N=256 CS=512 H=8 D=64 K=32 NB=65 CT=2048.
// GEMM: 128x128 block tile, 8 waves x (32x64), BK=32, register double
// buffering (global->VGPR prefetch of tile k+1 under WMMA of tile k),
// global_prefetch of tile k+2, XOR bank-swizzled LDS, packed dword stores.
// ---------------------------------------------------------------------------

typedef _Float16 half_t;
typedef __attribute__((ext_vector_type(16))) _Float16     v16h;
typedef __attribute__((ext_vector_type(8)))  float        v8f;
typedef __attribute__((ext_vector_type(4)))  unsigned int u32x4;
typedef __attribute__((ext_vector_type(4)))  float        f32x4;
typedef __attribute__((ext_vector_type(4)))  _Float16     h16x4;

union F16Frag {
    v16h         h;
    unsigned int u[8];
    u32x4        q[2];
};

__device__ __forceinline__ v8f wmma_f16(v16h a, v16h b, v8f c) {
    // (neg_a, A, neg_b, B, c_mod, C, reuse_a, reuse_b)
    return __builtin_amdgcn_wmma_f32_16x16x32_f16(false, a, false, b,
                                                  (short)0, c, false, false);
}

__device__ __forceinline__ unsigned int pack2h(float a, float b) {
    union { half_t h[2]; unsigned int u; } t;
    t.h[0] = (half_t)a; t.h[1] = (half_t)b;
    return t.u;
}

// Model constants
#define BB   8
#define NN   256
#define CSZ  512
#define HH   8
#define DD   64
#define NBIN 65
#define CT   2048
#define MROWS 2048           // B*N
#define KCAT 1032            // (NB+D)*H
#define KCATP 1056           // padded to multiple of 32
#define WLC  0.70710678118654752f
#define QSCALE (0.70710678118654752f / 8.0f)   // w_L / sqrt(D)

// ---------------------------------------------------------------------------
// Generic tiled WMMA GEMM: C[M,Nc] = A_f16[M,Ka] * W_f32[Ka,Nc]  (fp32 accum)
// ---------------------------------------------------------------------------
enum { EPI_HEADS = 0, EPI_HEADS_VT = 1, EPI_F32_BIAS = 2, EPI_F16_RELU = 3 };

template <int EPI>
__global__ __launch_bounds__(256) void gemm_wmma(
    const half_t* __restrict__ A, int lda,
    const float* __restrict__ W, int Nc, int kTiles, int kAct,
    const float* __restrict__ bias, float scale,
    void* __restrict__ outp)
{
    // [buf][row 0..127][dword 0..15], dword index XOR-swizzled by (row&15)
    __shared__ unsigned int sA[2][128 * 16];
    __shared__ unsigned int sB[2][128 * 16];

    const int tid  = threadIdx.x;
    const int lane = tid & 31;
    const int wid  = tid >> 5;
    const int wm   = wid & 3;        // 4 wave-rows of 32
    const int wn   = wid >> 2;       // 2 wave-cols of 64
    const int g    = lane >> 4;      // lane group (K-block select)
    const int lr   = lane & 15;      // M / N index within tile
    const int m0   = blockIdx.x * 128;
    const int n0   = blockIdx.y * 128;

    // per-thread staging coordinates
    const int aRow = tid >> 2;            // 0..63 (and +64 for second load)
    const int aQ   = tid & 3;             // which 16B chunk of the 64B row
    const int wKp  = tid >> 5;            // 0..7 : packed K-pair index
    const int wNn  = (tid & 31) * 4;      // 0,4,...,124 : 4 consecutive n

    u32x4 rA0, rA1;                       // A tile prefetch regs (2 x 16B)
    f32x4 rW0, rW1;                       // W tile prefetch regs (2 k-rows x 4n)

    auto loadTile = [&](int kt) {
        const int k0 = kt * 32;
        rA0 = *reinterpret_cast<const u32x4*>(
            A + (size_t)(m0 + aRow) * lda + k0 + aQ * 8);
        rA1 = *reinterpret_cast<const u32x4*>(
            A + (size_t)(m0 + aRow + 64) * lda + k0 + aQ * 8);
        const int kg0 = k0 + 2 * wKp, kg1 = kg0 + 1;
        f32x4 z = {0.0f, 0.0f, 0.0f, 0.0f};
        rW0 = (kg0 < kAct) ? *reinterpret_cast<const f32x4*>(
                                 W + (size_t)kg0 * Nc + n0 + wNn) : z;
        rW1 = (kg1 < kAct) ? *reinterpret_cast<const f32x4*>(
                                 W + (size_t)kg1 * Nc + n0 + wNn) : z;
    };
    auto storeTile = [&](int buf) {
#pragma unroll
        for (int c = 0; c < 4; ++c) {
            sA[buf][aRow * 16 + ((aQ * 4 + c) ^ (aRow & 15))]               = rA0[c];
            sA[buf][(aRow + 64) * 16 + ((aQ * 4 + c) ^ ((aRow + 64) & 15))] = rA1[c];
            sB[buf][(wNn + c) * 16 + (wKp ^ ((wNn + c) & 15))] =
                pack2h(rW0[c], rW1[c]);
        }
    };

    v8f acc[2][4] = {};

    loadTile(0);
    int buf = 0;
    for (int kt = 0; kt < kTiles; ++kt) {
        storeTile(buf);
        __syncthreads();

        if (kt + 1 < kTiles) loadTile(kt + 1);            // overlap with WMMA
        if (kt + 2 < kTiles) {                            // warm L2 two ahead
            const int k2 = (kt + 2) * 32;
            __builtin_prefetch(A + (size_t)(m0 + aRow) * lda + k2, 0, 3);
            if (k2 + 2 * wKp + 1 < kAct)
                __builtin_prefetch(W + (size_t)(k2 + 2 * wKp) * Nc + n0 + wNn,
                                   0, 3);
        }

        F16Frag af[2], bf[4];
#pragma unroll
        for (int i = 0; i < 2; ++i) {
            int rowA = wm * 32 + i * 16 + lr;
#pragma unroll
            for (int d = 0; d < 8; ++d) {   // A: K = g*16 + 2*d
                int kd2 = g * 8 + d;
                af[i].u[d] = sA[buf][rowA * 16 + (kd2 ^ lr)];
            }
        }
#pragma unroll
        for (int j = 0; j < 4; ++j) {
            int nn = wn * 64 + j * 16 + lr;
#pragma unroll
            for (int d = 0; d < 8; ++d)     // B: K = g*16 + 2*d
                bf[j].u[d] = sB[buf][nn * 16 + ((g * 8 + d) ^ lr)];
        }
#pragma unroll
        for (int i = 0; i < 2; ++i)
#pragma unroll
            for (int j = 0; j < 4; ++j)
                acc[i][j] = wmma_f16(af[i].h, bf[j].h, acc[i][j]);
        __syncthreads();
        buf ^= 1;
    }

    // ---- epilogue: C layout lane=(n, m-half), vgpr r: M = r + 8*(lane/16) ----
#pragma unroll
    for (int i = 0; i < 2; ++i)
#pragma unroll
        for (int j = 0; j < 4; ++j)
#pragma unroll
            for (int r = 0; r < 8; ++r) {
                int row = m0 + wm * 32 + i * 16 + r + 8 * g;
                int col = n0 + wn * 64 + j * 16 + lr;
                float v = acc[i][j][r] * scale;
                if constexpr (EPI == EPI_HEADS) {
                    // col = d*H + h ; out[b,h,n,d]
                    int b = row >> 8, n = row & 255, d = col >> 3, h = col & 7;
                    ((half_t*)outp)[((size_t)((b * HH + h) * NN + n)) * DD + d] =
                        (half_t)v;
                } else if constexpr (EPI == EPI_HEADS_VT) {
                    // out[b,h,d,n]  (transposed V for contiguous B-fragments)
                    int b = row >> 8, n = row & 255, d = col >> 3, h = col & 7;
                    ((half_t*)outp)[((size_t)((b * HH + h) * DD + d)) * NN + n] =
                        (half_t)v;
                } else if constexpr (EPI == EPI_F32_BIAS) {
                    ((float*)outp)[(size_t)row * Nc + col] = v + bias[col];
                } else { // EPI_F16_RELU
                    float y = v + bias[col];
                    ((half_t*)outp)[(size_t)row * Nc + col] =
                        (half_t)(y > 0.0f ? y : 0.0f);
                }
            }
}

// ---------------------------------------------------------------------------
// logits[b,h,i,j] = (WL/sqrt(D))*q.k (q pre-scaled) + WL*wb[bin,h]*sq - 1e9*(1-sq)
// Fragments loaded directly from global (K=64, L2-resident).
// Block 128 thr = 4 waves; wave tile 32x64; grid (B*H, N/64, N/128).
// ---------------------------------------------------------------------------
__global__ __launch_bounds__(128) void attn_logits_wmma(
    const half_t* __restrict__ qf, const half_t* __restrict__ kf,
    const float* __restrict__ wb, const unsigned char* __restrict__ masks,
    float* __restrict__ logits)
{
    const int tid = threadIdx.x, lane = tid & 31, wid = tid >> 5;
    const int g = lane >> 4, lr = lane & 15;
    const int wm = wid & 1, wn = wid >> 1;
    const int bh = blockIdx.x, b = bh >> 3, h = bh & 7;
    const int i0 = blockIdx.y * 64 + wm * 32;
    const int j0 = blockIdx.z * 128 + wn * 64;

    v8f acc[2][4] = {};
#pragma unroll
    for (int s = 0; s < 2; ++s) {
        F16Frag af[2], bf[4];
#pragma unroll
        for (int i = 0; i < 2; ++i) {
            const half_t* p = qf + ((size_t)bh * NN + (i0 + i * 16 + lr)) * DD;
            af[i].q[0] = *reinterpret_cast<const u32x4*>(p + s * 32 + g * 16);
            af[i].q[1] = *reinterpret_cast<const u32x4*>(p + s * 32 + g * 16 + 8);
        }
#pragma unroll
        for (int j = 0; j < 4; ++j) {
            const half_t* p = kf + ((size_t)bh * NN + (j0 + j * 16 + lr)) * DD;
            bf[j].q[0] = *reinterpret_cast<const u32x4*>(p + s * 32 + g * 16);
            bf[j].q[1] = *reinterpret_cast<const u32x4*>(p + s * 32 + g * 16 + 8);
        }
#pragma unroll
        for (int i = 0; i < 2; ++i)
#pragma unroll
            for (int j = 0; j < 4; ++j)
                acc[i][j] = wmma_f16(af[i].h, bf[j].h, acc[i][j]);
    }
#pragma unroll
    for (int i = 0; i < 2; ++i)
#pragma unroll
        for (int j = 0; j < 4; ++j)
#pragma unroll
            for (int r = 0; r < 8; ++r) {
                int ig = i0 + i * 16 + r + 8 * g;
                int jg = j0 + j * 16 + lr;
                int d  = ig - jg;
                d = d < -32 ? -32 : (d > 32 ? 32 : d);
                float mi = masks[b * NN + ig] ? 1.0f : 0.0f;
                float mj = masks[b * NN + jg] ? 1.0f : 0.0f;
                float sq = mi * mj;
                float v  = acc[i][j][r] + WLC * wb[(d + 32) * HH + h] * sq
                           - 1e9f * (1.0f - sq);
                logits[((size_t)bh * NN + ig) * NN + jg] = v;
            }
}

// ---------------------------------------------------------------------------
// Softmax: one wave per row of 256 (b128 vectorized); writes a (fp32, d_out)
// + f16 copy for the A.V WMMA.
// ---------------------------------------------------------------------------
__global__ __launch_bounds__(256) void softmax_rows(
    const float* __restrict__ logits, float* __restrict__ aout,
    half_t* __restrict__ a16)
{
    const int tid = threadIdx.x, lane = tid & 31, wid = tid >> 5;
    const size_t row = (size_t)blockIdx.x * 8 + wid;
    const f32x4* p4 = reinterpret_cast<const f32x4*>(logits + row * NN);
    f32x4 xa = p4[lane * 2], xb = p4[lane * 2 + 1];
    float m = -1e30f;
#pragma unroll
    for (int c = 0; c < 4; ++c) m = fmaxf(m, fmaxf(xa[c], xb[c]));
    for (int o = 16; o > 0; o >>= 1) m = fmaxf(m, __shfl_xor(m, o, 32));
    float s = 0.0f;
#pragma unroll
    for (int c = 0; c < 4; ++c) {
        xa[c] = __expf(xa[c] - m); s += xa[c];
        xb[c] = __expf(xb[c] - m); s += xb[c];
    }
    for (int o = 16; o > 0; o >>= 1) s += __shfl_xor(s, o, 32);
    float inv = 1.0f / s;
    xa *= inv; xb *= inv;
    f32x4* o4 = reinterpret_cast<f32x4*>(aout + row * NN);
    o4[lane * 2] = xa; o4[lane * 2 + 1] = xb;
    h16x4 ha, hb;
#pragma unroll
    for (int c = 0; c < 4; ++c) { ha[c] = (half_t)xa[c]; hb[c] = (half_t)xb[c]; }
    h16x4* h4 = reinterpret_cast<h16x4*>(a16 + row * NN);
    h4[lane * 2] = ha; h4[lane * 2 + 1] = hb;
}

// ---------------------------------------------------------------------------
// o_pair: bucket-sums of a over clipped relative-position bins (one-hot).
// Interior bins (1..63): single lookup j = i+32-c. Edge bins: wave-reduced
// partial sums. One wave per (b,h,i) row; writes f16 into concat[:, h*65+c].
// ---------------------------------------------------------------------------
__global__ __launch_bounds__(256) void opair_rows(
    const float* __restrict__ aout, const unsigned char* __restrict__ masks,
    half_t* __restrict__ concat)
{
    const int tid = threadIdx.x, lane = tid & 31, wid = tid >> 5;
    const size_t row = (size_t)blockIdx.x * 8 + wid;   // over B*H*N
    const int bh = (int)(row >> 8), i = (int)(row & 255);
    const int b = bh >> 3, h = bh & 7;
    const float* p = aout + row * NN;
    const float mi = masks[b * NN + i] ? 1.0f : 0.0f;

    float slo = 0.0f, shi = 0.0f;
    for (int j = lane; j < NN; j += 32) {
        float av = p[j] * (masks[b * NN + j] ? 1.0f : 0.0f);
        if (j >= i + 32) shi += av;     // bin 0  (i-j <= -K)
        if (j <= i - 32) slo += av;     // bin 64 (i-j >=  K)
    }
    for (int o = 16; o > 0; o >>= 1) { shi += __shfl_xor(shi, o, 32);
                                       slo += __shfl_xor(slo, o, 32); }
    half_t* dst = concat + ((size_t)(b * NN + i)) * KCATP + h * NBIN;
    for (int c = lane; c < NBIN; c += 32) {
        float v;
        if (c == 0)       v = shi;
        else if (c == 64) v = slo;
        else {
            int j = i + 32 - c;
            v = (j >= 0 && j < NN)
                    ? p[j] * (masks[b * NN + j] ? 1.0f : 0.0f) : 0.0f;
        }
        dst[c] = (half_t)(v * mi);
    }
}

// ---------------------------------------------------------------------------
// o = a @ v : per (b,h) GEMM [256x256]x[256x64], WMMA, V pre-transposed.
// Block 128 thr = 4 waves, wave tile 32 rows x 64 cols; grid (B*H, N/128).
// Writes f16 into concat[:, 520 + h*64 + d].
// ---------------------------------------------------------------------------
__global__ __launch_bounds__(128) void attn_av_wmma(
    const half_t* __restrict__ a16, const half_t* __restrict__ vt,
    half_t* __restrict__ concat)
{
    const int tid = threadIdx.x, lane = tid & 31, wid = tid >> 5;
    const int g = lane >> 4, lr = lane & 15;
    const int bh = blockIdx.x, b = bh >> 3, h = bh & 7;
    const int i0 = blockIdx.y * 128 + wid * 32;

    v8f acc[2][4] = {};
#pragma unroll
    for (int ks = 0; ks < 8; ++ks) {
        F16Frag af[2], bf[4];
#pragma unroll
        for (int i = 0; i < 2; ++i) {
            const half_t* p = a16 + ((size_t)bh * NN + (i0 + i * 16 + lr)) * NN;
            af[i].q[0] = *reinterpret_cast<const u32x4*>(p + ks * 32 + g * 16);
            af[i].q[1] = *reinterpret_cast<const u32x4*>(p + ks * 32 + g * 16 + 8);
        }
#pragma unroll
        for (int j = 0; j < 4; ++j) {
            const half_t* p = vt + ((size_t)bh * DD + (j * 16 + lr)) * NN;
            bf[j].q[0] = *reinterpret_cast<const u32x4*>(p + ks * 32 + g * 16);
            bf[j].q[1] = *reinterpret_cast<const u32x4*>(p + ks * 32 + g * 16 + 8);
        }
#pragma unroll
        for (int i = 0; i < 2; ++i)
#pragma unroll
            for (int j = 0; j < 4; ++j)
                acc[i][j] = wmma_f16(af[i].h, bf[j].h, acc[i][j]);
    }
#pragma unroll
    for (int i = 0; i < 2; ++i)
#pragma unroll
        for (int j = 0; j < 4; ++j)
#pragma unroll
            for (int r = 0; r < 8; ++r) {
                int ig = i0 + i * 16 + r + 8 * g;
                int dn = j * 16 + lr;
                concat[((size_t)(b * NN + ig)) * KCATP + 520 + h * DD + dn] =
                    (half_t)acc[i][j][r];
            }
}

// ---------------------------------------------------------------------------
// x = LayerNorm(base + delta) : one wave per 512-wide row (b128 vectorized).
// ---------------------------------------------------------------------------
__global__ __launch_bounds__(256) void add_layernorm(
    const float* __restrict__ base, const float* __restrict__ delta,
    const float* __restrict__ gamma, const float* __restrict__ beta,
    float* __restrict__ out32, half_t* __restrict__ out16)
{
    const int tid = threadIdx.x, lane = tid & 31, wid = tid >> 5;
    const size_t row = (size_t)blockIdx.x * 8 + wid;   // 2048 rows
    const f32x4* pb4 = reinterpret_cast<const f32x4*>(base + row * CSZ);
    const f32x4* pd4 = reinterpret_cast<const f32x4*>(delta + row * CSZ);
    f32x4 x[4];
    float s1 = 0.0f, s2 = 0.0f;
#pragma unroll
    for (int q = 0; q < 4; ++q) {
        x[q] = pb4[lane * 4 + q] + pd4[lane * 4 + q];
#pragma unroll
        for (int c = 0; c < 4; ++c) { s1 += x[q][c]; s2 += x[q][c] * x[q][c]; }
    }
    for (int o = 16; o > 0; o >>= 1) { s1 += __shfl_xor(s1, o, 32);
                                       s2 += __shfl_xor(s2, o, 32); }
    float mean = s1 * (1.0f / CSZ);
    float var  = s2 * (1.0f / CSZ) - mean * mean;
    float rs   = rsqrtf(var + 1e-5f);
    const f32x4* g4 = reinterpret_cast<const f32x4*>(gamma);
    const f32x4* b4 = reinterpret_cast<const f32x4*>(beta);
    f32x4* o4 = reinterpret_cast<f32x4*>(out32 + row * CSZ);
    h16x4* h4 = out16 ? reinterpret_cast<h16x4*>(out16 + row * CSZ) : nullptr;
#pragma unroll
    for (int q = 0; q < 4; ++q) {
        f32x4 y = (x[q] - mean) * rs * g4[lane * 4 + q] + b4[lane * 4 + q];
        o4[lane * 4 + q] = y;
        if (h4) {
            h16x4 hy;
#pragma unroll
            for (int c = 0; c < 4; ++c) hy[c] = (half_t)y[c];
            h4[lane * 4 + q] = hy;
        }
    }
}

__global__ void f32_to_f16(const float* __restrict__ in,
                           half_t* __restrict__ out, int n4)
{
    const f32x4* in4 = reinterpret_cast<const f32x4*>(in);
    h16x4* out4 = reinterpret_cast<h16x4*>(out);
    for (int i = blockIdx.x * blockDim.x + threadIdx.x; i < n4;
         i += gridDim.x * blockDim.x) {
        f32x4 v = in4[i];
        h16x4 h;
#pragma unroll
        for (int c = 0; c < 4; ++c) h[c] = (half_t)v[c];
        out4[i] = h;
    }
}

// ---------------------------------------------------------------------------
// Workspace layout (bytes; logits region reused for FFN-hidden, embd for ff)
// ---------------------------------------------------------------------------
static constexpr size_t OFF_S16    = 0;                         // 2 MB
static constexpr size_t OFF_Q16    = OFF_S16 + 2097152;         // 2 MB
static constexpr size_t OFF_K16    = OFF_Q16 + 2097152;         // 2 MB
static constexpr size_t OFF_VT16   = OFF_K16 + 2097152;         // 2 MB
static constexpr size_t OFF_LOGITS = OFF_VT16 + 2097152;        // 16 MB (reused as h_f16: 8 MB)
static constexpr size_t OFF_A16    = OFF_LOGITS + 16777216;     // 8 MB
static constexpr size_t OFF_CONCAT = OFF_A16 + 8388608;         // 2048*1056*2 = 4.125 MB
static constexpr size_t OFF_EMBD   = OFF_CONCAT + (size_t)MROWS * KCATP * 2; // 4 MB (reused as ff)
static constexpr size_t OFF_X      = OFF_EMBD + 4194304;        // 4 MB
static constexpr size_t OFF_X16    = OFF_X + 4194304;           // 2 MB
// total ~46.1 MiB

extern "C" void kernel_launch(void* const* d_in, const int* in_sizes, int n_in,
                              void* d_out, int out_size, void* d_ws, size_t ws_size,
                              hipStream_t stream)
{
    (void)in_sizes; (void)n_in; (void)out_size; (void)ws_size;

    const float*         s     = (const float*)d_in[0];
    const unsigned char* masks = (const unsigned char*)d_in[1];
    const float* wq = (const float*)d_in[2];
    const float* wk = (const float*)d_in[3];
    const float* wv = (const float*)d_in[4];
    const float* wb = (const float*)d_in[5];
    const float* wo = (const float*)d_in[6];
    const float* bo = (const float*)d_in[7];
    const float* g1 = (const float*)d_in[8];
    const float* b1n = (const float*)d_in[9];
    const float* w1 = (const float*)d_in[10];
    const float* bb1 = (const float*)d_in[11];
    const float* w2 = (const float*)d_in[12];
    const float* bb2 = (const float*)d_in[13];
    const float* g2 = (const float*)d_in[14];
    const float* b2n = (const float*)d_in[15];

    char* ws = (char*)d_ws;
    half_t* s16    = (half_t*)(ws + OFF_S16);
    half_t* q16    = (half_t*)(ws + OFF_Q16);
    half_t* k16    = (half_t*)(ws + OFF_K16);
    half_t* vt16   = (half_t*)(ws + OFF_VT16);
    float*  logits = (float*)(ws + OFF_LOGITS);
    half_t* h16    = (half_t*)(ws + OFF_LOGITS);   // reuse after softmax
    half_t* a16    = (half_t*)(ws + OFF_A16);
    half_t* concat = (half_t*)(ws + OFF_CONCAT);
    float*  embd   = (float*)(ws + OFF_EMBD);
    float*  ffout  = (float*)(ws + OFF_EMBD);      // reuse after first LN
    float*  x      = (float*)(ws + OFF_X);
    half_t* x16    = (half_t*)(ws + OFF_X16);

    float* xout = (float*)d_out;                       // [B,N,CS]
    float* aout = (float*)d_out + (size_t)MROWS * CSZ; // [B,H,N,N]

    // 1) s -> f16
    f32_to_f16<<<512, 256, 0, stream>>>(s, s16, MROWS * CSZ / 4);

    // 2-4) QKV projections (q pre-scaled by w_L/sqrt(D); v stored transposed)
    gemm_wmma<EPI_HEADS><<<dim3(16, 4), 256, 0, stream>>>(
        s16, CSZ, wq, CSZ, 16, CSZ, nullptr, QSCALE, q16);
    gemm_wmma<EPI_HEADS><<<dim3(16, 4), 256, 0, stream>>>(
        s16, CSZ, wk, CSZ, 16, CSZ, nullptr, 1.0f, k16);
    gemm_wmma<EPI_HEADS_VT><<<dim3(16, 4), 256, 0, stream>>>(
        s16, CSZ, wv, CSZ, 16, CSZ, nullptr, 1.0f, vt16);

    // 5) logits = scaled QK^T + rel-pos bias + mask
    attn_logits_wmma<<<dim3(BB * HH, 4, 2), 128, 0, stream>>>(
        q16, k16, wb, masks, logits);

    // 6) softmax -> a (fp32 to d_out) + f16 copy
    softmax_rows<<<2048, 256, 0, stream>>>(logits, aout, a16);

    // 7) concat buffer: zero (covers the 1032..1055 K-padding), then fill
    hipMemsetAsync(concat, 0, (size_t)MROWS * KCATP * 2, stream);
    opair_rows<<<2048, 256, 0, stream>>>(aout, masks, concat);
    attn_av_wmma<<<dim3(BB * HH, 2), 128, 0, stream>>>(a16, vt16, concat);

    // 8) output projection: embd = concat[2048x1032] @ wo + bo
    gemm_wmma<EPI_F32_BIAS><<<dim3(16, 4), 256, 0, stream>>>(
        concat, KCATP, wo, CSZ, KCATP / 32, KCAT, bo, 1.0f, embd);

    // 9) x = LN(s + embd)
    add_layernorm<<<256, 256, 0, stream>>>(s, embd, g1, b1n, x, x16);

    // 10) FFN: h = relu(x@w1+b1) (f16), ff = h@w2+b2 (fp32)
    gemm_wmma<EPI_F16_RELU><<<dim3(16, 16), 256, 0, stream>>>(
        x16, CSZ, w1, CT, 16, CSZ, bb1, 1.0f, h16);
    gemm_wmma<EPI_F32_BIAS><<<dim3(16, 4), 256, 0, stream>>>(
        h16, CT, w2, CSZ, 64, CT, bb2, 1.0f, ffout);

    // 11) out = LN(x + ff)
    add_layernorm<<<256, 256, 0, stream>>>(x, ffout, g2, b2n, xout, nullptr);
}
